// PatchedCausalSelfAttention_45320494907534
// MI455X (gfx1250) — matrix operand order repaired and checked by reading, not compiled, not run on hardware
//
#include <hip/hip_runtime.h>
#include <hip/hip_bf16.h>

// ---------------------------------------------------------------------------
// Causal self-attention (GPT-2) B=4, S=2048, D=1024, H=16, hd=64 on gfx1250.
// Heavy math: V_WMMA_F32_16X16X32_BF16. GEMM tiles staged by the Tensor Data
// Mover (tensor_load_to_lds, TENSORcnt double-buffered). Weights pre-
// transposed and V stored [B,H,hd,S] so all fragment loads are contiguous
// 32-bit pairs (-> ds_load_b128 / global_load_b128, no u16 merge chains).
// ---------------------------------------------------------------------------

typedef __attribute__((ext_vector_type(16))) __bf16 v16bf;
typedef __attribute__((ext_vector_type(8)))  float  v8f;
typedef __attribute__((ext_vector_type(4)))  unsigned tdm_v4u;
typedef __attribute__((ext_vector_type(8)))  int      tdm_v8i;
typedef __attribute__((ext_vector_type(4)))  int      tdm_v4i;

union FragA { v16bf v; unsigned u[8]; };   // 16x32 bf16 A/B fragment
union FragC { v8f   v; float    f[8]; };   // 16x16 f32  C/D fragment

#define WMMA_BF16(A, B, C) \
  __builtin_amdgcn_wmma_f32_16x16x32_bf16(false, (A), false, (B), (short)0, (C), false, false)

#if __has_builtin(__builtin_amdgcn_tensor_load_to_lds)
#define HAVE_TDM 1
#else
#define HAVE_TDM 0
#endif
#if __has_include(<hip/amd_detail/amd_gfx1250_TDM.h>)
#define TDM_6ARG 1
#else
#define TDM_6ARG 0
#endif

static constexpr int Bb   = 4;
static constexpr int S    = 2048;
static constexpr int D    = 1024;
static constexpr int H    = 16;
static constexpr int HD   = 64;
static constexpr int M    = Bb * S;          // 8192 rows
static constexpr int NH_E = M * D;
static constexpr int NA_E = D * 3 * D;
static constexpr int NP_E = D * D;

// ---------------------------------------------------------------------------
// fp32 -> bf16 cast (plain)
// ---------------------------------------------------------------------------
__global__ void cvt_f32_bf16(const float* __restrict__ in, __bf16* __restrict__ out, int n) {
  int i = blockIdx.x * blockDim.x + threadIdx.x;
  if (i < n) out[i] = (__bf16)in[i];
}

// fp32 [K x NC] -> bf16 transposed [NC x K] (weight packing)
__global__ void cvt_t_f32_bf16(const float* __restrict__ in, __bf16* __restrict__ out,
                               int K, int NC) {
  const int n = blockIdx.x * blockDim.x + threadIdx.x;
  const int k = blockIdx.y;
  if (n < NC) out[(size_t)n * K + k] = (__bf16)in[(size_t)k * NC + n];
}

#if HAVE_TDM
// One 2-D TDM tile load: tile_d1 rows of tile_d0 bf16 elements, row stride
// d0_stride elements, into LDS at byte offset lds_off. D# per CDNA5 ISA
// CDNA5 ISA (groups 0/1; groups 2/3 zero; tile_dim2=0 unused).
__device__ __forceinline__ void tdm_load_2d(unsigned lds_off, const void* gaddr,
                                            unsigned tile_d0, unsigned tile_d1,
                                            unsigned long long d0_stride) {
  unsigned long long ga = (unsigned long long)gaddr;
  tdm_v4u g0;
  g0[0] = 1u;                                            // count=1, user desc
  g0[1] = lds_off;                                       // lds_addr (bytes)
  g0[2] = (unsigned)(ga & 0xFFFFFFFFull);                // global_addr[31:0]
  g0[3] = (unsigned)((ga >> 32) & 0x1FFFFFFull) | (2u << 30);  // addr[56:32] | type=2
  tdm_v8i g1;
  g1[0] = (int)(1u << 16);                               // data_size=1 (2 bytes)
  g1[1] = (int)((tile_d0 & 0xFFFFu) << 16);              // tensor_dim0 = tile_d0
  g1[2] = (int)((tile_d0 >> 16) | ((tile_d1 & 0xFFFFu) << 16));  // tensor_dim1
  g1[3] = (int)((tile_d1 >> 16) | (tile_d0 << 16));      // tile_dim0
  g1[4] = (int)(tile_d1 & 0xFFFFu);                      // tile_dim1, tile_dim2=0
  g1[5] = (int)(d0_stride & 0xFFFFFFFFull);              // dim0_stride lo
  g1[6] = (int)((d0_stride >> 32) & 0xFFFFull);          // dim0_stride hi
  g1[7] = 0;
  tdm_v4i gz = {0, 0, 0, 0};
#if TDM_6ARG
  tdm_v8i gz8 = {0, 0, 0, 0, 0, 0, 0, 0};
  __builtin_amdgcn_tensor_load_to_lds(g0, g1, gz, gz, gz8, 0);
#else
  __builtin_amdgcn_tensor_load_to_lds(g0, g1, gz, gz, 0);
#endif
}
#endif

// ---------------------------------------------------------------------------
// GEMM: C[M x NC] = A[M x 1024](bf16) @ Wt[NC x 1024](bf16, pre-transposed)
// + bias.  256 threads = 8 waves; block tile 128x128; wave tile 32x64
// (2x4 subtiles); K-tile 64, double-buffered TDM staging. Both LDS tiles are
// k-contiguous ([row][64] / [col][64]) so fragment reads are ds_load_b128.
// mode 0: scatter bf16 Q/K [B,H,S,hd], V [B,H,hd,S];  mode 1: f32 out.
// ---------------------------------------------------------------------------
__global__ __launch_bounds__(256)
void gemm_bf16(const __bf16* __restrict__ A, const __bf16* __restrict__ Wt,
               const float* __restrict__ bias, int NC, int mode,
               __bf16* __restrict__ qb, __bf16* __restrict__ kb, __bf16* __restrict__ vt,
               float* __restrict__ outf) {
  __shared__ __bf16 lds_a[2][128 * 64];   // [row][k]  2 x 16 KB
  __shared__ __bf16 lds_b[2][128 * 64];   // [col][k]  2 x 16 KB

  const int tid  = threadIdx.x;
  const int bm   = blockIdx.y;            // 128-row tile
  const int bn   = blockIdx.x;            // 128-col tile
  const int wave = tid >> 5;
  const int lane = tid & 31;
  const int lm   = lane & 15;
  const int lh   = lane >> 4;
  const int wr   = wave >> 1;             // 0..3
  const int wc   = wave & 1;              // 0..1

  FragC acc[2][4];
#pragma unroll
  for (int i = 0; i < 2; ++i)
#pragma unroll
    for (int j = 0; j < 4; ++j)
#pragma unroll
      for (int v = 0; v < 8; ++v) acc[i][j].f[v] = 0.0f;

  constexpr int NK = D / 64;              // 16 K-tiles

  auto stage = [&](int kt, int buf) {
#if HAVE_TDM
    if (wave == 0) {
      tdm_load_2d((unsigned)(size_t)&lds_a[buf][0],
                  A + (size_t)(bm * 128) * D + kt * 64, 64, 128, (unsigned long long)D);
      tdm_load_2d((unsigned)(size_t)&lds_b[buf][0],
                  Wt + (size_t)(bn * 128) * D + kt * 64, 64, 128, (unsigned long long)D);
    }
#else
    {  // fallback: cooperative staged copy (both operands k-contiguous)
      const int r = tid >> 1, half = tid & 1;
      const uint4* ga = (const uint4*)(A + (size_t)(bm * 128 + r) * D + kt * 64 + 32 * half);
      uint4* la = (uint4*)&lds_a[buf][r * 64 + 32 * half];
#pragma unroll
      for (int q = 0; q < 4; ++q) la[q] = ga[q];
      const uint4* gb = (const uint4*)(Wt + (size_t)(bn * 128 + r) * D + kt * 64 + 32 * half);
      uint4* lb = (uint4*)&lds_b[buf][r * 64 + 32 * half];
#pragma unroll
      for (int q = 0; q < 4; ++q) lb[q] = gb[q];
    }
#endif
  };

  stage(0, 0);
  for (int kt = 0; kt < NK; ++kt) {
    const int buf = kt & 1;
    if (kt + 1 < NK) stage(kt + 1, buf ^ 1);
#if HAVE_TDM
    if (wave == 0) {
      if (kt + 1 < NK) __builtin_amdgcn_s_wait_tensorcnt(1);  // tile kt done (in-order)
      else             __builtin_amdgcn_s_wait_tensorcnt(0);
    }
#endif
    __syncthreads();

    const unsigned* la32 = (const unsigned*)lds_a[buf];
    const unsigned* lb32 = (const unsigned*)lds_b[buf];
#pragma unroll
    for (int s = 0; s < 2; ++s) {           // two K=32 steps per LDS tile
      FragA af[2], bfr[4];
#pragma unroll
      for (int i = 0; i < 2; ++i) {
        const int row = wr * 32 + 16 * i + lm;
#pragma unroll
        for (int v = 0; v < 8; ++v) {
          const int kbase = s * 32 + 16 * lh + 2 * v;  // A layout
          af[i].u[v] = la32[(row * 64 + kbase) >> 1];
        }
      }
#pragma unroll
      for (int j = 0; j < 4; ++j) {
        const int col = wc * 64 + 16 * j + lm;
#pragma unroll
        for (int v = 0; v < 8; ++v) {
          const int kbase = s * 32 + 16 * lh + 2 * v;                           // B layout
          bfr[j].u[v] = lb32[(col * 64 + kbase) >> 1];
        }
      }
#pragma unroll
      for (int i = 0; i < 2; ++i)
#pragma unroll
        for (int j = 0; j < 4; ++j)
          acc[i][j].v = WMMA_BF16(af[i].v, bfr[j].v, acc[i][j].v);
    }
    __syncthreads();
  }

  // --- epilogue: C/D layout m = v + 8*lh, n = lm ---
#pragma unroll
  for (int i = 0; i < 2; ++i)
#pragma unroll
    for (int j = 0; j < 4; ++j)
#pragma unroll
      for (int v = 0; v < 8; ++v) {
        const int gm = bm * 128 + wr * 32 + 16 * i + v + 8 * lh;
        const int gn = bn * 128 + wc * 64 + 16 * j + lm;
        const float val = acc[i][j].f[v] + bias[gn];
        if (mode == 0) {
          const int which = gn >> 10;           // 0=q 1=k 2=v
          const int d = gn & 1023;
          const int h = d >> 6, hd = d & 63;
          const int b = gm >> 11, sq = gm & 2047;
          if (which == 2) {                     // V transposed: [B,H,hd,S]
            vt[(((size_t)(b * H + h)) * HD + hd) * S + sq] = (__bf16)val;
          } else {
            const size_t idx = (((size_t)(b * H + h)) * S + sq) * HD + hd;
            ((which == 0) ? qb : kb)[idx] = (__bf16)val;
          }
        } else {
          outf[(size_t)gm * NC + gn] = val;
        }
      }
}

// ---------------------------------------------------------------------------
// Flash attention: one wave per (b, h, 16-query tile); online softmax over
// 32-key chunks; row stats at m = v + 8*(lane/16) match the WMMA C layout;
// P bounced through LDS (s_wait_dscnt) to repack D-layout -> A-layout.
// V is [B,H,hd,S] so PV B-fragments are contiguous u32 loads.
// ---------------------------------------------------------------------------
__global__ __launch_bounds__(128)
void attn_flash(const __bf16* __restrict__ qb, const __bf16* __restrict__ kb,
                const __bf16* __restrict__ vt, __bf16* __restrict__ ob) {
  __shared__ __bf16 lds_p[4 * 16 * 32];

  const int tid  = threadIdx.x;
  const int wave = tid >> 5;
  const int lane = tid & 31;
  const int lm   = lane & 15;
  const int lh   = lane >> 4;

  const int unit = blockIdx.x * 4 + wave;
  const int qt   = unit & 127;
  const int bh   = unit >> 7;
  const int q0   = qt * 16;

  const size_t base = (size_t)bh * S * HD;
  const unsigned* q32 = (const unsigned*)(qb + base);
  const unsigned* k32 = (const unsigned*)(kb + base);
  const unsigned* v32 = (const unsigned*)(vt + base);   // [hd][S]

  FragA qa[2];
  {
    const int qrow = (q0 + lm) * HD;
#pragma unroll
    for (int f = 0; f < 2; ++f)
#pragma unroll
      for (int v = 0; v < 8; ++v) {
        const int kbase = 32 * f + 16 * lh + 2 * v;
        qa[f].u[v] = q32[(qrow + kbase) >> 1];
      }
  }

  float mrun[8], lrun[8];
  FragC o[4];
#pragma unroll
  for (int v = 0; v < 8; ++v) { mrun[v] = -1e30f; lrun[v] = 0.0f; }
#pragma unroll
  for (int cc = 0; cc < 4; ++cc)
#pragma unroll
    for (int v = 0; v < 8; ++v) o[cc].f[v] = 0.0f;

  const int nch = (q0 + 47) >> 5;
  for (int c = 0; c < nch; ++c) {
    const int keys0 = c * 32;

    FragC s[2];
#pragma unroll
    for (int t = 0; t < 2; ++t) {
      const int key = keys0 + 16 * t + lm;
      const unsigned* krow = k32 + (size_t)key * (HD / 2);
      FragA kb0, kb1;
#pragma unroll
      for (int v = 0; v < 8; ++v) {
        kb0.u[v] = krow[8 * lh + v];
        kb1.u[v] = krow[8 * lh + v + 16];
      }
      v8f z = {0.f, 0.f, 0.f, 0.f, 0.f, 0.f, 0.f, 0.f};
      s[t].v = WMMA_BF16(qa[0].v, kb0.v, z);
      s[t].v = WMMA_BF16(qa[1].v, kb1.v, s[t].v);
#pragma unroll
      for (int v = 0; v < 8; ++v) {
        float val = s[t].f[v] * 0.125f;
        const int qrow = q0 + v + 8 * lh;
        if (key > qrow) val = -10000.0f;
        s[t].f[v] = val;
      }
    }

    float mx[8], ef[8], sum[8];
#pragma unroll
    for (int v = 0; v < 8; ++v) mx[v] = fmaxf(s[0].f[v], s[1].f[v]);
#pragma unroll
    for (int off = 8; off >= 1; off >>= 1)
#pragma unroll
      for (int v = 0; v < 8; ++v) mx[v] = fmaxf(mx[v], __shfl_xor(mx[v], off, 32));
#pragma unroll
    for (int v = 0; v < 8; ++v) {
      const float mnew = fmaxf(mrun[v], mx[v]);
      ef[v] = __expf(mrun[v] - mnew);
      mrun[v] = mnew;
    }
#pragma unroll
    for (int t = 0; t < 2; ++t)
#pragma unroll
      for (int v = 0; v < 8; ++v) s[t].f[v] = __expf(s[t].f[v] - mrun[v]);
#pragma unroll
    for (int v = 0; v < 8; ++v) sum[v] = s[0].f[v] + s[1].f[v];
#pragma unroll
    for (int off = 8; off >= 1; off >>= 1)
#pragma unroll
      for (int v = 0; v < 8; ++v) sum[v] += __shfl_xor(sum[v], off, 32);
#pragma unroll
    for (int v = 0; v < 8; ++v) lrun[v] = lrun[v] * ef[v] + sum[v];
#pragma unroll
    for (int cc = 0; cc < 4; ++cc)
#pragma unroll
      for (int v = 0; v < 8; ++v) o[cc].f[v] *= ef[v];

    __bf16* pw = lds_p + wave * 512;
#pragma unroll
    for (int t = 0; t < 2; ++t)
#pragma unroll
      for (int v = 0; v < 8; ++v) {
        const int m = v + 8 * lh;
        pw[m * 32 + 16 * t + lm] = (__bf16)s[t].f[v];
      }
    asm volatile("s_wait_dscnt 0" ::: "memory");
    FragA pa;
    {
      const unsigned* p32 = (const unsigned*)pw;
#pragma unroll
      for (int v = 0; v < 8; ++v) {
        const int kbase = 16 * lh + 2 * v;
        pa.u[v] = p32[(lm * 32 + kbase) >> 1];
      }
    }

#pragma unroll
    for (int cc = 0; cc < 4; ++cc) {
      const int hd = cc * 16 + lm;                        // this lane's hd column
      const unsigned* vrow = v32 + (size_t)hd * (S / 2) + (keys0 >> 1);
      FragA vf;
#pragma unroll
      for (int v = 0; v < 8; ++v) vf.u[v] = vrow[8 * lh + v];   // keys contiguous
      o[cc].v = WMMA_BF16(pa.v, vf.v, o[cc].v);
    }
  }

  const int b = bh >> 4, h = bh & 15;
#pragma unroll
  for (int cc = 0; cc < 4; ++cc)
#pragma unroll
    for (int v = 0; v < 8; ++v) {
      const int row = q0 + v + 8 * lh;
      const int col = h * HD + cc * 16 + lm;
      ob[((size_t)b * S + row) * D + col] = (__bf16)(o[cc].f[v] / lrun[v]);
    }
}

// ---------------------------------------------------------------------------
// launch
// ---------------------------------------------------------------------------
extern "C" void kernel_launch(void* const* d_in, const int* in_sizes, int n_in,
                              void* d_out, int out_size, void* d_ws, size_t ws_size,
                              hipStream_t stream) {
  const float* hs  = (const float*)d_in[0];
  const float* Wa  = (const float*)d_in[1];
  const float* ba  = (const float*)d_in[2];
  const float* Wp  = (const float*)d_in[3];
  const float* bp  = (const float*)d_in[4];
  float*       out = (float*)d_out;

  char* w = (char*)d_ws;
  __bf16* hsb   = (__bf16*)w; w += (size_t)NH_E * 2;
  __bf16* wabT  = (__bf16*)w; w += (size_t)NA_E * 2;   // [3072][1024]
  __bf16* wpbT  = (__bf16*)w; w += (size_t)NP_E * 2;   // [1024][1024]
  __bf16* qb    = (__bf16*)w; w += (size_t)NH_E * 2;
  __bf16* kbuf  = (__bf16*)w; w += (size_t)NH_E * 2;
  __bf16* vtb   = (__bf16*)w; w += (size_t)NH_E * 2;   // [B,H,hd,S]
  __bf16* attnb = (__bf16*)w; w += (size_t)NH_E * 2;   // ~88 MB total

  cvt_f32_bf16<<<(NH_E + 255) / 256, 256, 0, stream>>>(hs, hsb, NH_E);
  cvt_t_f32_bf16<<<dim3(3 * D / 256, D), 256, 0, stream>>>(Wa, wabT, D, 3 * D);
  cvt_t_f32_bf16<<<dim3(D / 256, D), 256, 0, stream>>>(Wp, wpbT, D, D);

  gemm_bf16<<<dim3(3 * D / 128, M / 128), 256, 0, stream>>>(
      hsb, wabT, ba, 3 * D, /*mode=*/0, qb, kbuf, vtb, nullptr);

  attn_flash<<<(Bb * H * (S / 16)) / 4, 128, 0, stream>>>(qb, kbuf, vtb, attnb);

  gemm_bf16<<<dim3(D / 128, M / 128), 256, 0, stream>>>(
      attnb, wpbT, bp, D, /*mode=*/1, nullptr, nullptr, nullptr, out);
}
